// Attention_18605798326350
// MI455X (gfx1250) — compile-verified, matrix operand order checked by reading, not run on hardware
//
#include <hip/hip_runtime.h>
#include <stdint.h>

typedef __attribute__((ext_vector_type(16))) __bf16 v16bf;
typedef __attribute__((ext_vector_type(8)))  float  v8f;
typedef int v4i_vs __attribute__((vector_size(16)));   // matches builtin param

#define DEV_INLINE __device__ __forceinline__

constexpr int Bc = 2, Sc = 2048, DIMc = 4096, Hc = 32, KVHc = 8, HDc = 128, NREP = 4;

// ---- CDNA5 async Global->LDS copy (ASYNCcnt path), with fallback ----------
#if defined(__has_builtin)
#if __has_builtin(__builtin_amdgcn_global_load_async_to_lds_b128)
#define HAVE_ASYNC_LDS 1
#endif
#endif

DEV_INLINE void cp16(void* l, const void* g) {
#if defined(HAVE_ASYNC_LDS)
  __builtin_amdgcn_global_load_async_to_lds_b128(
      (v4i_vs*)g, (v4i_vs*)l, 0, 0);
#else
  *reinterpret_cast<uint4*>(l) = *reinterpret_cast<const uint4*>(g);
#endif
}

DEV_INLINE void wait_async() {
#if defined(HAVE_ASYNC_LDS)
#if __has_builtin(__builtin_amdgcn_s_wait_asynccnt)
  __builtin_amdgcn_s_wait_asynccnt(0);
#else
  asm volatile("s_wait_asynccnt 0x0" ::: "memory");
#endif
#endif
}

DEV_INLINE unsigned short f32_to_bf16(float f) {
  union { float f; unsigned u; } c; c.f = f;
  unsigned u = c.u + 0x7FFFu + ((c.u >> 16) & 1u);   // round-to-nearest-even
  return (unsigned short)(u >> 16);
}

// A/B fragment loader for V_WMMA_*_16x16x32 bf16 operands stored K-major.
// p points at element [row_or_col][khalf*8]; lane gathers K {base..base+7} and
// {base+16..base+23} per the ISA 16-bit A-matrix layout (section 7.12.2).
DEV_INLINE v16bf frag_ld16(const unsigned short* p) {
  union { v16bf v; unsigned short u[16]; } r;
#pragma unroll
  for (int i = 0; i < 8; ++i) { r.u[i] = p[i]; r.u[i + 8] = p[i + 16]; }
  return r.v;
}

// Same fragment but the source is [K][N]-major (stride ld between K rows).
DEV_INLINE v16bf frag_ld16_strided(const unsigned short* p, int ld) {
  union { v16bf v; unsigned short u[16]; } r;
#pragma unroll
  for (int i = 0; i < 8; ++i) { r.u[i] = p[i * ld]; r.u[i + 8] = p[(i + 16) * ld]; }
  return r.v;
}

// ---------------------------------------------------------------- convert ---
__global__ void cvt_f32_to_bf16(const float4* __restrict__ in,
                                ushort4* __restrict__ out, long n4) {
  long i = (long)blockIdx.x * blockDim.x + threadIdx.x;
  if (i >= n4) return;
  float4 v = in[i];
  ushort4 o;
  o.x = f32_to_bf16(v.x); o.y = f32_to_bf16(v.y);
  o.z = f32_to_bf16(v.z); o.w = f32_to_bf16(v.w);
  out[i] = o;
}

// ------------------------------------------------------------------- GEMM ---
// C[M,N] (f32) = A[M,K] (bf16, row-major) * W[N,K]^T (bf16, row-major rows)
#define BM 128
#define BN 128
#define BKK 64

__global__ __launch_bounds__(256) void gemm_bf16_wmma(
    const unsigned short* __restrict__ A,
    const unsigned short* __restrict__ W,
    float* __restrict__ C,
    int M, int N, int K)
{
  __shared__ __align__(16) unsigned short As[2][BM][BKK];   // 2 x 16 KB
  __shared__ __align__(16) unsigned short Ws[2][BN][BKK];   // 2 x 16 KB

  const int tid   = threadIdx.x;
  const int lane  = tid & 31;
  const int wave  = tid >> 5;        // 8 waves
  const int wm    = wave >> 1;       // 0..3 : 32 rows each
  const int wn    = wave & 1;        // 0..1 : 64 cols each
  const int lrow  = lane & 15;
  const int khalf = lane >> 4;

  const int bm = blockIdx.y * BM;
  const int bn = blockIdx.x * BN;

  v8f acc[2][4] = {};

  // stage one BMxBKK / BNxBKK tile pair into LDS buffer `buf`
  auto stage = [&](int k0, int buf) {
#pragma unroll
    for (int j = 0; j < 4; ++j) {                 // 1024 uint4 / 256 threads
      int idx = tid + j * 256;
      int row = idx >> 3;                         // 8 uint4 per 64-elem row
      int col = (idx & 7) * 8;
      cp16(&As[buf][row][col], A + (long)(bm + row) * K + k0 + col);
      cp16(&Ws[buf][row][col], W + (long)(bn + row) * K + k0 + col);
    }
  };

  stage(0, 0);
  wait_async();
  __syncthreads();

  const int nk = K / BKK;
  for (int t = 0; t < nk; ++t) {
    const int cur = t & 1;
    if (t + 1 < nk) stage((t + 1) * BKK, cur ^ 1);   // overlap with compute

#pragma unroll
    for (int kk = 0; kk < BKK; kk += 32) {
      v16bf af[2], bfr[4];
#pragma unroll
      for (int tm = 0; tm < 2; ++tm)
        af[tm] = frag_ld16(&As[cur][wm * 32 + tm * 16 + lrow][kk + khalf * 8]);
#pragma unroll
      for (int tn = 0; tn < 4; ++tn)
        bfr[tn] = frag_ld16(&Ws[cur][wn * 64 + tn * 16 + lrow][kk + khalf * 8]);
#pragma unroll
      for (int tm = 0; tm < 2; ++tm)
#pragma unroll
        for (int tn = 0; tn < 4; ++tn)
          acc[tm][tn] = __builtin_amdgcn_wmma_f32_16x16x32_bf16(
              false, af[tm], false, bfr[tn], (short)0, acc[tm][tn], false, false);
    }

    wait_async();
    __syncthreads();
  }

#pragma unroll
  for (int tm = 0; tm < 2; ++tm) {
#pragma unroll
    for (int tn = 0; tn < 4; ++tn) {
      int rowbase = bm + wm * 32 + tm * 16 + khalf * 8;
      int col     = bn + wn * 64 + tn * 16 + lrow;
#pragma unroll
      for (int i = 0; i < 8; ++i)
        C[(long)(rowbase + i) * N + col] = acc[tm][tn][i];
    }
  }
}

// ------------------------------------------------------------------- RoPE ---
// in: f32 [B,S,heads,HD], out: bf16 same layout, one thread per (re,im) pair
__global__ void rope_f32_to_bf16(const float* __restrict__ in,
                                 const float* __restrict__ fc,
                                 const float* __restrict__ fs,
                                 unsigned short* __restrict__ out,
                                 int heads, long total_pairs)
{
  long i = (long)blockIdx.x * blockDim.x + threadIdx.x;
  if (i >= total_pairs) return;
  int p = (int)(i % (HDc / 2));
  long rest = i / (HDc / 2);
  long bs = rest / heads;
  int s = (int)(bs % Sc);
  float c  = fc[s * (HDc / 2) + p];
  float sn = fs[s * (HDc / 2) + p];
  float re = in[i * 2], im = in[i * 2 + 1];
  out[i * 2]     = f32_to_bf16(re * c - im * sn);
  out[i * 2 + 1] = f32_to_bf16(re * sn + im * c);
}

// -------------------------------------------------------- flash attention ---
#define QT 64
#define KT 64

__global__ __launch_bounds__(128) void flash_attn_wmma(
    const unsigned short* __restrict__ Q,     // [B,S,H,HD] bf16 (roped)
    const unsigned short* __restrict__ Kmat,  // [B,S,KVH,HD] bf16 (roped)
    const unsigned short* __restrict__ Vmat,  // [B,S,KVH,HD] bf16
    unsigned short* __restrict__ O,           // [B,S,H*HD] bf16
    float scale)
{
  __shared__ __align__(16) unsigned short Ks[2][KT][HDc];   // 2 x 16 KB
  __shared__ __align__(16) unsigned short Vs[2][KT][HDc];   // 2 x 16 KB
  __shared__ __align__(16) unsigned short Ps[4][16][KT];    //     8 KB

  const int tid   = threadIdx.x;
  const int lane  = tid & 31;
  const int w     = tid >> 5;      // 4 waves: 16 query rows each
  const int lrow  = lane & 15;
  const int khalf = lane >> 4;

  const int q0 = blockIdx.x * QT;
  const int h  = blockIdx.y;
  const int b  = blockIdx.z;
  const int hk = h / NREP;

  auto stage_kv = [&](int kt, int buf) {
#pragma unroll
    for (int j = 0; j < 8; ++j) {        // 1024 uint4 per tile / 128 threads
      int idx = tid + j * 128;
      int row = idx >> 4;
      int col = (idx & 15) * 8;
      long gk = (((long)(b * Sc + kt * KT + row) * KVHc) + hk) * HDc + col;
      cp16(&Ks[buf][row][col], Kmat + gk);
      cp16(&Vs[buf][row][col], Vmat + gk);
    }
  };

  // Q fragments resident in VGPRs: 4 k-steps of 32 over HD=128
  v16bf aq[4];
  {
    int qrow = q0 + w * 16 + lrow;
    const unsigned short* qp = Q + (((long)(b * Sc + qrow) * Hc) + h) * HDc;
#pragma unroll
    for (int kq = 0; kq < 4; ++kq)
      aq[kq] = frag_ld16(qp + kq * 32 + khalf * 8);
  }

  v8f   o_acc[8] = {};             // 16 rows x 128 hd
  float m_s[8], l_s[8];
#pragma unroll
  for (int i = 0; i < 8; ++i) { m_s[i] = -3.0e38f; l_s[i] = 0.0f; }

  const int ktiles = blockIdx.x + 1;     // causal: keys <= q0+QT-1
  stage_kv(0, 0);
  wait_async();
  __syncthreads();

  for (int kt = 0; kt < ktiles; ++kt) {
    const int cur = kt & 1;
    if (kt + 1 < ktiles) stage_kv(kt + 1, cur ^ 1);   // overlap with compute

    // S = Q K^T : 4 key tiles x 4 k-steps
    v8f s[4] = {};
#pragma unroll
    for (int nt = 0; nt < 4; ++nt)
#pragma unroll
      for (int kq = 0; kq < 4; ++kq) {
        v16bf bk = frag_ld16(&Ks[cur][nt * 16 + lrow][kq * 32 + khalf * 8]);
        s[nt] = __builtin_amdgcn_wmma_f32_16x16x32_bf16(
            false, aq[kq], false, bk, (short)0, s[nt], false, false);
      }

    // scale + causal mask (C layout: row = i + khalf*8, col = lane&15)
    const int qbase = q0 + w * 16 + khalf * 8;
#pragma unroll
    for (int nt = 0; nt < 4; ++nt) {
      int key = kt * KT + nt * 16 + lrow;
#pragma unroll
      for (int i = 0; i < 8; ++i) {
        float v = s[nt][i] * scale;
        if (key > qbase + i) v = -1.0e9f;
        s[nt][i] = v;
      }
    }

    // online softmax: per-row max over 16-lane groups (wave32 shuffles)
    float m_new[8], alpha[8];
#pragma unroll
    for (int i = 0; i < 8; ++i) {
      float v = fmaxf(fmaxf(s[0][i], s[1][i]), fmaxf(s[2][i], s[3][i]));
#pragma unroll
      for (int off = 8; off >= 1; off >>= 1)
        v = fmaxf(v, __shfl_xor(v, off, 32));
      m_new[i] = fmaxf(m_s[i], v);
      alpha[i] = __expf(m_s[i] - m_new[i]);
      m_s[i]   = m_new[i];
    }

    float rsum[8];
#pragma unroll
    for (int i = 0; i < 8; ++i) rsum[i] = 0.0f;
#pragma unroll
    for (int nt = 0; nt < 4; ++nt)
#pragma unroll
      for (int i = 0; i < 8; ++i) {
        float pv = __expf(s[nt][i] - m_new[i]);
        rsum[i] += pv;
        Ps[w][i + khalf * 8][nt * 16 + lrow] = f32_to_bf16(pv);
      }
#pragma unroll
    for (int i = 0; i < 8; ++i) {
      float v = rsum[i];
#pragma unroll
      for (int off = 8; off >= 1; off >>= 1)
        v += __shfl_xor(v, off, 32);
      l_s[i] = l_s[i] * alpha[i] + v;
    }
#pragma unroll
    for (int t = 0; t < 8; ++t)
#pragma unroll
      for (int i = 0; i < 8; ++i)
        o_acc[t][i] *= alpha[i];

    // O += P V : 2 k-steps of 32 keys, 8 hd tiles
#pragma unroll
    for (int kp = 0; kp < 2; ++kp) {
      v16bf ap = frag_ld16(&Ps[w][lrow][kp * 32 + khalf * 8]);
#pragma unroll
      for (int nt = 0; nt < 8; ++nt) {
        v16bf bv = frag_ld16_strided(&Vs[cur][kp * 32 + khalf * 8][nt * 16 + lrow], HDc);
        o_acc[nt] = __builtin_amdgcn_wmma_f32_16x16x32_bf16(
            false, ap, false, bv, (short)0, o_acc[nt], false, false);
      }
    }

    wait_async();
    __syncthreads();
  }

  // normalize + store bf16 [B,S,H*HD]
#pragma unroll
  for (int nt = 0; nt < 8; ++nt)
#pragma unroll
    for (int i = 0; i < 8; ++i) {
      int qrow = q0 + w * 16 + khalf * 8 + i;
      float v = o_acc[nt][i] / l_s[i];
      O[((long)(b * Sc + qrow) * DIMc) + h * HDc + nt * 16 + lrow] = f32_to_bf16(v);
    }
}

// ----------------------------------------------------------------- launch ---
extern "C" void kernel_launch(void* const* d_in, const int* in_sizes, int n_in,
                              void* d_out, int out_size, void* d_ws, size_t ws_size,
                              hipStream_t stream) {
  (void)in_sizes; (void)n_in; (void)out_size; (void)ws_size;
  const float* x  = (const float*)d_in[0];
  const float* wq = (const float*)d_in[1];
  const float* wk = (const float*)d_in[2];
  const float* wv = (const float*)d_in[3];
  const float* wo = (const float*)d_in[4];
  const float* fc = (const float*)d_in[5];
  const float* fs = (const float*)d_in[6];
  // d_in[7] = mask (causal mask computed inline), d_in[8] = start_pos (0)
  float* out = (float*)d_out;

  const long BS = (long)Bc * Sc;        // 4096
  const long KV = (long)KVHc * HDc;     // 1024

  char* p = (char*)d_ws;
  auto take = [&](long bytes) { char* r = p; p += (bytes + 255) & ~255L; return r; };

  unsigned short* xb  = (unsigned short*)take(BS * DIMc * 2);
  unsigned short* wqb = (unsigned short*)take((long)DIMc * DIMc * 2);
  unsigned short* wkb = (unsigned short*)take(KV * DIMc * 2);
  unsigned short* wvb = (unsigned short*)take(KV * DIMc * 2);
  unsigned short* wob = (unsigned short*)take((long)DIMc * DIMc * 2);
  float*          qf  = (float*)take(BS * DIMc * 4);
  float*          kf  = (float*)take(BS * KV * 4);
  float*          vf  = (float*)take(BS * KV * 4);
  unsigned short* qb  = (unsigned short*)take(BS * DIMc * 2);
  unsigned short* kb  = (unsigned short*)take(BS * KV * 2);
  unsigned short* vb  = (unsigned short*)take(BS * KV * 2);
  unsigned short* ab  = (unsigned short*)take(BS * DIMc * 2);

  auto cvt = [&](const float* src, unsigned short* dst, long n) {
    long n4 = n / 4;
    cvt_f32_to_bf16<<<(n4 + 255) / 256, 256, 0, stream>>>(
        (const float4*)src, (ushort4*)dst, n4);
  };
  cvt(x,  xb,  BS * DIMc);
  cvt(wq, wqb, (long)DIMc * DIMc);
  cvt(wk, wkb, KV * DIMc);
  cvt(wv, wvb, KV * DIMc);
  cvt(wo, wob, (long)DIMc * DIMc);

  // QKV projections (f32 accum out)
  gemm_bf16_wmma<<<dim3(DIMc / BN, BS / BM), 256, 0, stream>>>(xb, wqb, qf, (int)BS, DIMc, DIMc);
  gemm_bf16_wmma<<<dim3((int)KV / BN, BS / BM), 256, 0, stream>>>(xb, wkb, kf, (int)BS, (int)KV, DIMc);
  gemm_bf16_wmma<<<dim3((int)KV / BN, BS / BM), 256, 0, stream>>>(xb, wvb, vf, (int)BS, (int)KV, DIMc);

  // RoPE (+ bf16 cast) for Q and K; plain cast for V
  long qpairs = BS * Hc * (HDc / 2);
  rope_f32_to_bf16<<<(qpairs + 255) / 256, 256, 0, stream>>>(qf, fc, fs, qb, Hc, qpairs);
  long kpairs = BS * KVHc * (HDc / 2);
  rope_f32_to_bf16<<<(kpairs + 255) / 256, 256, 0, stream>>>(kf, fc, fs, kb, KVHc, kpairs);
  cvt(vf, vb, BS * KV);

  // flash attention (GQA handled by hk = h / NREP)
  const float scale = 0.08838834764831845f;   // 1/sqrt(128)
  flash_attn_wmma<<<dim3(Sc / QT, Hc, Bc), 128, 0, stream>>>(qb, kb, vb, ab, scale);

  // output projection straight into d_out (f32)
  gemm_bf16_wmma<<<dim3(DIMc / BN, BS / BM), 256, 0, stream>>>(ab, wob, out, (int)BS, DIMc, DIMc);
}
